// AttentionDecoder_3264175145068
// MI455X (gfx1250) — matrix operand order repaired and checked by reading, not compiled, not run on hardware
//
#include <hip/hip_runtime.h>
#include <cstdint>
#include <cstddef>

typedef __bf16 bf16_t;
typedef __attribute__((ext_vector_type(16))) __bf16 v16bf;
typedef __attribute__((ext_vector_type(8)))  __bf16 v8bf;
typedef __attribute__((ext_vector_type(8)))  float  v8f;

constexpr int B = 64, T = 32, S = 64, H = 1024, V = 32000;
constexpr int H2 = 2 * H, H3 = 3 * H;

// d_out layout (floats): decoder_output | decoder_hidden | attention_weights | ctx_fin
constexpr size_t OFF_HID = (size_t)B * T * V;
constexpr size_t OFF_ATT = OFF_HID + (size_t)B * H;
constexpr size_t OFF_CTX = OFF_ATT + (size_t)B * T * S;

__device__ __forceinline__ bf16_t f2bf(float f) {
  uint32_t u = __builtin_bit_cast(uint32_t, f);
  u = (u + 0x7FFFu + ((u >> 16) & 1u)) >> 16;   // round-to-nearest-even
  uint16_t s = (uint16_t)u;
  return __builtin_bit_cast(bf16_t, s);
}

__global__ void k_cvt_bf16(const float* __restrict__ src, bf16_t* __restrict__ dst, int n) {
  int i = blockIdx.x * 256 + threadIdx.x;
  if (i < n) dst[i] = f2bf(src[i]);
}

__global__ void k_init(const float* __restrict__ mem, float* __restrict__ h,
                       float* __restrict__ ctx, bf16_t* __restrict__ cat) {
  int i = blockIdx.x * 256 + threadIdx.x;
  if (i >= B * H) return;
  int b = i / H, j = i % H;
  float v = mem[i];                 // memory[0] is (B,H)
  h[i] = v;
  ctx[i] = 0.0f;
  cat[(size_t)b * H2 + H + j] = f2bf(v);   // h0 (bf16) lives in cat[:, H:2H]
}

__global__ void k_build_x(const int* __restrict__ input, int t,
                          const float* __restrict__ emb, const float* __restrict__ ctx,
                          bf16_t* __restrict__ x) {
  int i = blockIdx.x * 256 + threadIdx.x;
  if (i >= B * H2) return;
  int b = i / H2, c = i % H2;
  float v = (c < H) ? emb[(size_t)input[b * T + t] * H + c]
                    : ctx[(size_t)b * H + (c - H)];
  x[i] = f2bf(v);
}

// C[M,N] (f32, ldc) = A[M,K] (bf16, row-major, lda) * W[N,K]^T (bf16, row-major) + bias[N]
// Register-blocked: each wave computes an (MT x NT) grid of 16x16 tiles, so each
// K-step issues MT*NT WMMAs from MT+NT fragment loads (B-fragments reused MT times).
template<int MT, int NT>
__global__ void __launch_bounds__(256)
k_gemm_bf16(const bf16_t* __restrict__ A, int lda,
            const bf16_t* __restrict__ W,
            const float* __restrict__ bias,
            float* __restrict__ C, int ldc,
            int M, int N, int K) {
  int wave = blockIdx.x * 8 + (threadIdx.x >> 5);
  int nBlocksN = N / (16 * NT);
  int total = (M / (16 * MT)) * nBlocksN;
  if (wave >= total) return;                     // wave-uniform exit, EXEC stays full
  int bm = wave / nBlocksN, bn = wave % nBlocksN;
  int lane = threadIdx.x & 31;
  int half = lane >> 4, lm = lane & 15;

  const bf16_t* arow[MT];
#pragma unroll
  for (int mt = 0; mt < MT; ++mt)
    arow[mt] = A + (size_t)((bm * MT + mt) * 16 + lm) * lda;
  const bf16_t* wrow[NT];
#pragma unroll
  for (int nt = 0; nt < NT; ++nt)
    wrow[nt] = W + (size_t)((bn * NT + nt) * 16 + lm) * K;

  v8f acc[MT][NT];
#pragma unroll
  for (int mt = 0; mt < MT; ++mt)
#pragma unroll
    for (int nt = 0; nt < NT; ++nt)
      acc[mt][nt] = (v8f){0.f, 0.f, 0.f, 0.f, 0.f, 0.f, 0.f, 0.f};

  union U { v16bf v; v8bf h[2]; };
  for (int k0 = 0; k0 < K; k0 += 32) {
    U a[MT], b[NT];
#pragma unroll
    for (int mt = 0; mt < MT; ++mt) {
      // A 16-bit 16x32 layout: lanes<16 hold K 0..7 & 16..23; lanes>=16 hold 8..15 & 24..31
      a[mt].h[0] = *(const v8bf*)(arow[mt] + k0 + 8 * half);
      a[mt].h[1] = *(const v8bf*)(arow[mt] + k0 + 16 + 8 * half);
    }
#pragma unroll
    for (int nt = 0; nt < NT; ++nt) {
      // B 16-bit 32x16 layout: lanes<16 hold K 0..7 & 16..23 (N=lm); lanes>=16 hold 8..15 & 24..31
      b[nt].h[0] = *(const v8bf*)(wrow[nt] + k0 + 8 * half);
      b[nt].h[1] = *(const v8bf*)(wrow[nt] + k0 + 16 + 8 * half);
    }
#pragma unroll
    for (int mt = 0; mt < MT; ++mt)
#pragma unroll
      for (int nt = 0; nt < NT; ++nt)
        acc[mt][nt] = __builtin_amdgcn_wmma_f32_16x16x32_bf16(
            false, a[mt].v, false, b[nt].v, (short)0, acc[mt][nt], false, false);
  }

#pragma unroll
  for (int nt = 0; nt < NT; ++nt) {
    int n = (bn * NT + nt) * 16 + lm;
    float bv = bias ? bias[n] : 0.0f;
#pragma unroll
    for (int mt = 0; mt < MT; ++mt) {
#pragma unroll
      for (int j = 0; j < 8; ++j) {              // D: m = 8*half + j, n = lm
        int m = (bm * MT + mt) * 16 + half * 8 + j;
        C[(size_t)m * ldc + n] = acc[mt][nt][j] + bv;
      }
    }
  }
}

__global__ void k_gru_fuse(const float* __restrict__ gi, const float* __restrict__ gh,
                           float* __restrict__ h, bf16_t* __restrict__ cat) {
  int i = blockIdx.x * 256 + threadIdx.x;
  if (i >= B * H) return;
  int b = i / H, j = i % H;
  const float* gib = gi + (size_t)b * H3;
  const float* ghb = gh + (size_t)b * H3;
  float r = 1.0f / (1.0f + expf(-(gib[j] + ghb[j])));
  float z = 1.0f / (1.0f + expf(-(gib[H + j] + ghb[H + j])));
  float n = tanhf(gib[2 * H + j] + r * ghb[2 * H + j]);
  float hn = (1.0f - z) * n + z * h[i];
  h[i] = hn;
  cat[(size_t)b * H2 + H + j] = f2bf(hn);         // hidden half of concat (also A for q-gemm)
}

__global__ void __launch_bounds__(256)
k_attention(const float* __restrict__ enc, const int* __restrict__ xs_len,
            const float* __restrict__ q, bf16_t* __restrict__ cat,
            float* __restrict__ attn_out, int t) {
  int b = blockIdx.x, tid = threadIdx.x;
  __shared__ float spart[256];
  __shared__ float sval[S];
  int s = tid & (S - 1);
  int chunk = tid >> 6;                            // 4 chunks of H/4
  const float* erow = enc + ((size_t)b * S + s) * H;
  const float* qrow = q + (size_t)b * H;
  float p = 0.0f;
  int base = chunk * (H / 4);
  for (int i = 0; i < H / 4; ++i) p += erow[base + i] * qrow[base + i];
  spart[tid] = p;
  __syncthreads();
  if (tid < S) {
    float sc = spart[tid] + spart[tid + 64] + spart[tid + 128] + spart[tid + 192];
    float m = (tid < xs_len[b]) ? sc : 0.0f;       // seq_mask * score
    sval[tid] = (m == 0.0f) ? -1e10f : m;          // masked==0 -> -1e10 (ref quirk)
  }
  __syncthreads();
  if (tid == 0) {                                  // softmax over 64 values, serial is fine
    float mx = -3.4e38f;
    for (int k = 0; k < S; ++k) mx = fmaxf(mx, sval[k]);
    float sum = 0.0f;
    for (int k = 0; k < S; ++k) { float e = expf(sval[k] - mx); sval[k] = e; sum += e; }
    float inv = 1.0f / sum;
    for (int k = 0; k < S; ++k) sval[k] *= inv;
  }
  __syncthreads();
  if (tid < S) attn_out[((size_t)b * T + t) * S + tid] = sval[tid];
  for (int h0 = tid; h0 < H; h0 += 256) {          // ctx = attn @ enc -> cat[:, :H] (bf16)
    float acc = 0.0f;
    for (int k = 0; k < S; ++k) acc += sval[k] * enc[((size_t)b * S + k) * H + h0];
    cat[(size_t)b * H2 + h0] = f2bf(acc);
  }
}

__global__ void k_tanh_ctx(const float* __restrict__ ctx2, float* __restrict__ ctx,
                           bf16_t* __restrict__ ctx_bf) {
  int i = blockIdx.x * 256 + threadIdx.x;
  if (i >= B * H) return;
  float v = tanhf(ctx2[i]);
  ctx[i] = v;
  ctx_bf[i] = f2bf(v);
}

__global__ void __launch_bounds__(256)
k_logsoftmax(const float* __restrict__ logits, float* __restrict__ out, int t) {
  int b = blockIdx.x, tid = threadIdx.x;
  const float* row = logits + (size_t)b * V;
  float* orow = out + ((size_t)b * T + t) * V;
  __shared__ float red[256];
  float mx = -3.4e38f;
  for (int v = tid; v < V; v += 256) mx = fmaxf(mx, row[v]);
  red[tid] = mx; __syncthreads();
  for (int st = 128; st > 0; st >>= 1) { if (tid < st) red[tid] = fmaxf(red[tid], red[tid + st]); __syncthreads(); }
  mx = red[0]; __syncthreads();
  float sum = 0.0f;
  for (int v = tid; v < V; v += 256) sum += expf(row[v] - mx);
  red[tid] = sum; __syncthreads();
  for (int st = 128; st > 0; st >>= 1) { if (tid < st) red[tid] += red[tid + st]; __syncthreads(); }
  float lse = mx + logf(red[0]);
  for (int v = tid; v < V; v += 256) orow[v] = row[v] - lse;
}

__global__ void k_final(const float* __restrict__ h, const float* __restrict__ ctx,
                        float* __restrict__ out) {
  int i = blockIdx.x * 256 + threadIdx.x;
  if (i >= B * H) return;
  out[OFF_HID + i] = h[i];
  out[OFF_CTX + i] = ctx[i];
}

extern "C" void kernel_launch(void* const* d_in, const int* in_sizes, int n_in,
                              void* d_out, int out_size, void* d_ws, size_t ws_size,
                              hipStream_t stream) {
  const int*   input  = (const int*)  d_in[0];
  const float* memory = (const float*)d_in[1];
  const float* enc    = (const float*)d_in[2];
  const int*   xs_len = (const int*)  d_in[3];
  const float* emb    = (const float*)d_in[4];
  const float* w_ih   = (const float*)d_in[5];
  const float* w_hh   = (const float*)d_in[6];
  const float* b_ih   = (const float*)d_in[7];
  const float* b_hh   = (const float*)d_in[8];
  const float* l1     = (const float*)d_in[9];
  const float* l2     = (const float*)d_in[10];
  const float* w_out  = (const float*)d_in[11];
  const float* b_out  = (const float*)d_in[12];
  float* out = (float*)d_out;

  char* ws = (char*)d_ws;
  size_t off = 0;
  auto take = [&](size_t bytes) -> char* {
    char* p = ws + off;
    off += (bytes + 255) & ~(size_t)255;
    return p;
  };
  bf16_t* wout_bf = (bf16_t*)take((size_t)V * H * 2);    // 65.5 MB, L2-resident
  bf16_t* wih_bf  = (bf16_t*)take((size_t)H3 * H2 * 2);
  bf16_t* whh_bf  = (bf16_t*)take((size_t)H3 * H * 2);
  bf16_t* l1_bf   = (bf16_t*)take((size_t)H * H * 2);
  bf16_t* l2_bf   = (bf16_t*)take((size_t)H * H2 * 2);
  bf16_t* x_bf    = (bf16_t*)take((size_t)B * H2 * 2);
  bf16_t* cat_bf  = (bf16_t*)take((size_t)B * H2 * 2);   // [ctx_attn | h] bf16
  bf16_t* ctx_bf  = (bf16_t*)take((size_t)B * H * 2);
  float*  hbuf    = (float*) take((size_t)B * H * 4);
  float*  ctx     = (float*) take((size_t)B * H * 4);
  float*  gi      = (float*) take((size_t)B * H3 * 4);
  float*  gh      = (float*) take((size_t)B * H3 * 4);
  float*  q       = (float*) take((size_t)B * H * 4);
  float*  ctx2    = (float*) take((size_t)B * H * 4);
  float*  logits  = (float*) take((size_t)B * V * 4);

  auto cdiv = [](int a, int b) { return (a + b - 1) / b; };

  k_cvt_bf16<<<cdiv(V * H, 256), 256, 0, stream>>>(w_out, wout_bf, V * H);
  k_cvt_bf16<<<cdiv(H3 * H2, 256), 256, 0, stream>>>(w_ih, wih_bf, H3 * H2);
  k_cvt_bf16<<<cdiv(H3 * H, 256), 256, 0, stream>>>(w_hh, whh_bf, H3 * H);
  k_cvt_bf16<<<cdiv(H * H, 256), 256, 0, stream>>>(l1, l1_bf, H * H);
  k_cvt_bf16<<<cdiv(H * H2, 256), 256, 0, stream>>>(l2, l2_bf, H * H2);
  k_init<<<cdiv(B * H, 256), 256, 0, stream>>>(memory, hbuf, ctx, cat_bf);

  // Small, latency-critical GEMMs: MT=1, NT=2 (max wave parallelism, 2x A-reuse).
  auto gemm_small = [&](const bf16_t* A, int lda, const bf16_t* W, const float* bias,
                        float* C, int ldc, int M, int N, int K) {
    int waves = (M / 16) * (N / 32);
    k_gemm_bf16<1, 2><<<cdiv(waves, 8), 256, 0, stream>>>(A, lda, W, bias, C, ldc, M, N, K);
  };
  // Big output projection: MT=4 (all of M), NT=2 -> 8 WMMAs per 6 fragment loads,
  // w_out streamed from L2 exactly once per step.
  auto gemm_big = [&](const bf16_t* A, int lda, const bf16_t* W, const float* bias,
                      float* C, int ldc, int M, int N, int K) {
    int waves = (M / 64) * (N / 32);
    k_gemm_bf16<4, 2><<<cdiv(waves, 8), 256, 0, stream>>>(A, lda, W, bias, C, ldc, M, N, K);
  };

  for (int t = 0; t < T; ++t) {
    k_build_x<<<cdiv(B * H2, 256), 256, 0, stream>>>(input, t, emb, ctx, x_bf);
    gemm_small(x_bf, H2, wih_bf, b_ih, gi, H3, B, H3, H2);        // gi = x @ w_ih^T
    gemm_small(cat_bf + H, H2, whh_bf, b_hh, gh, H3, B, H3, H);   // gh = h_prev @ w_hh^T
    k_gru_fuse<<<cdiv(B * H, 256), 256, 0, stream>>>(gi, gh, hbuf, cat_bf);
    gemm_small(cat_bf + H, H2, l1_bf, nullptr, q, H, B, H, H);    // q = h @ l1^T
    k_attention<<<B, 256, 0, stream>>>(enc, xs_len, q, cat_bf, out + OFF_ATT, t);
    gemm_small(cat_bf, H2, l2_bf, nullptr, ctx2, H, B, H, H2);    // [ctx,h] @ l2^T
    k_tanh_ctx<<<cdiv(B * H, 256), 256, 0, stream>>>(ctx2, ctx, ctx_bf);
    gemm_big(ctx_bf, H, wout_bf, b_out, logits, V, B, V, H);      // logits
    k_logsoftmax<<<B, 256, 0, stream>>>(logits, out, t);
  }
  k_final<<<cdiv(B * H, 256), 256, 0, stream>>>(hbuf, ctx, out);

  (void)in_sizes; (void)n_in; (void)out_size; (void)ws_size;
}
